// EncoderBiRNN_20504173871242
// MI455X (gfx1250) — matrix operand order repaired and checked by reading, not compiled, not run on hardware
//
#include <hip/hip_runtime.h>

typedef __bf16 bf16;
typedef __attribute__((ext_vector_type(8)))  bf16  v8bf;
typedef __attribute__((ext_vector_type(16))) bf16  v16bf;
typedef __attribute__((ext_vector_type(8)))  float v8f;

// ---------------- device helpers ----------------

__device__ __forceinline__ bf16 f2bf(float f) {
    unsigned u = __builtin_bit_cast(unsigned, f);
    unsigned r = u + 0x7FFFu + ((u >> 16) & 1u);
    unsigned short h = (unsigned short)(r >> 16);
    return __builtin_bit_cast(bf16, h);
}

__device__ __forceinline__ float sigm(float x) { return 1.0f / (1.0f + __expf(-x)); }

// A fragment (16x32 bf16, row-major, ld elements per row).
// Layout used here (not the ISA table order): lane m=lane&15 holds row M=m; half=lane>>4 selects K sub-blocks:
// elements 0..15 -> K = half*16 + 0..15 (contiguous, matching B)
__device__ __forceinline__ v16bf load_fragA(const bf16* base, int ld, int k0, int lane) {
    int m = lane & 15, half = lane >> 4;
    const bf16* p = base + (size_t)m * ld + k0 + half * 16;
    v8bf lo = *reinterpret_cast<const v8bf*>(p);
    v8bf hi = *reinterpret_cast<const v8bf*>(p + 8);
    return __builtin_shufflevector(lo, hi, 0,1,2,3,4,5,6,7,8,9,10,11,12,13,14,15);
}

// B fragment (32x16 bf16). Memory holds W[n, k] row-major -> column n contiguous in k.
// Layout used here (not the ISA table order): lane n=lane&15 holds column N=n; half selects K=half*16 + 0..15 (contiguous).
__device__ __forceinline__ v16bf load_fragB(const bf16* base, int ld, int k0, int lane) {
    int n = lane & 15, half = lane >> 4;
    const bf16* p = base + (size_t)n * ld + k0 + half * 16;
    return *reinterpret_cast<const v16bf*>(p);
}

__device__ __forceinline__ v8f wmma_bf16(v16bf a, v16bf b, v8f c) {
    return __builtin_amdgcn_wmma_f32_16x16x32_bf16(false, a, false, b, (short)0, c, false, false);
}

__device__ __forceinline__ v8f vzero8() {
    v8f a;
#pragma unroll
    for (int i = 0; i < 8; ++i) a[i] = 0.0f;
    return a;
}

// ---------------- utility kernels ----------------

__global__ void k_zero(float* p, long n) {
    long i = (long)blockIdx.x * blockDim.x + threadIdx.x;
    if (i < n) p[i] = 0.0f;
}

__global__ void k_f2bf(const float* in, bf16* out, long n) {
    long i = (long)blockIdx.x * blockDim.x + threadIdx.x;
    if (i < n) out[i] = f2bf(in[i]);
}

__global__ void k_embed(const int* tok, const float* emb, bf16* out, long total, int E) {
    long i = (long)blockIdx.x * blockDim.x + threadIdx.x;
    if (i >= total) return;
    long row = i / E;
    int  e   = (int)(i % E);
    int  t   = tok[row];
    out[i] = f2bf(emb[(size_t)t * E + e]);
}

// ---------------- WMMA GEMM: C[M,2048] = A[M,K](bf16) @ B[2048,K]^T(bf16) + bias ----------------
// grid = (M/16, 4), block = 256 (8 waves).
// Block stages its contiguous 16xK A tile into LDS once; wave w owns 4 column
// tiles (64 cols), reusing each A fragment across 4 WMMAs.
__global__ void k_gemm_bias(const bf16* __restrict__ A, const bf16* __restrict__ B,
                            const float* __restrict__ bias, float* __restrict__ C, int K) {
    __shared__ __attribute__((aligned(32))) bf16 As[16 * 1024];
    int tid = threadIdx.x, lane = tid & 31, wave = tid >> 5;
    int row0 = blockIdx.x * 16;
    // cooperative copy: 16 rows x K, contiguous since lda == K
    {
        const v8bf* src = reinterpret_cast<const v8bf*>(A + (size_t)row0 * K);
        v8bf* dst = reinterpret_cast<v8bf*>(As);
        int nv = (16 * K) >> 3;
        for (int i = tid; i < nv; i += 256) dst[i] = src[i];
    }
    __syncthreads();
    int colBase = blockIdx.y * 512 + wave * 64;
    const bf16* bp0 = B + (size_t)(colBase +  0) * K;
    const bf16* bp1 = B + (size_t)(colBase + 16) * K;
    const bf16* bp2 = B + (size_t)(colBase + 32) * K;
    const bf16* bp3 = B + (size_t)(colBase + 48) * K;
    v8f a0 = vzero8(), a1 = vzero8(), a2 = vzero8(), a3 = vzero8();
    for (int k0 = 0; k0 < K; k0 += 32) {
        v16bf a = load_fragA(As, K, k0, lane);
        a0 = wmma_bf16(a, load_fragB(bp0, K, k0, lane), a0);
        a1 = wmma_bf16(a, load_fragB(bp1, K, k0, lane), a1);
        a2 = wmma_bf16(a, load_fragB(bp2, K, k0, lane), a2);
        a3 = wmma_bf16(a, load_fragB(bp3, K, k0, lane), a3);
    }
    int half = lane >> 4, nn = lane & 15;
#pragma unroll
    for (int t = 0; t < 4; ++t) {
        v8f acc = (t == 0) ? a0 : (t == 1) ? a1 : (t == 2) ? a2 : a3;
        int col = colBase + t * 16 + nn;
        float bv = bias[col];
#pragma unroll
        for (int r = 0; r < 8; ++r) {
            int m = half * 8 + r;
            C[(size_t)(row0 + m) * 2048 + col] = acc[r] + bv;
        }
    }
}

// ---------------- fused LSTM timestep ----------------
// G = xg_t + h_prev @ Whh^T ; i,f,g,o gates ; c,h update.
// grid = (N/16, 4), block = 256. Block stages its 16x512 h tile into LDS.
// Wave w owns hidden cols [blockIdx.y*128 + w*16, +16) and computes all four
// gate tiles for those columns -> no cross-wave traffic for the cell update.
__global__ void k_lstm_step(const float* __restrict__ xgT, int xgStride,
                            const bf16* __restrict__ whh,
                            const bf16* __restrict__ hprev, bf16* __restrict__ hnext,
                            float* __restrict__ cst,
                            float* __restrict__ houtT, int houtStride) {
    __shared__ __attribute__((aligned(32))) bf16 Hs[16 * 512];
    int tid = threadIdx.x, lane = tid & 31, wave = tid >> 5;
    int row0  = blockIdx.x * 16;
    {
        const v8bf* src = reinterpret_cast<const v8bf*>(hprev + (size_t)row0 * 512);
        v8bf* dst = reinterpret_cast<v8bf*>(Hs);
        for (int i = tid; i < (16 * 512) / 8; i += 256) dst[i] = src[i];
    }
    __syncthreads();
    int hcol0 = blockIdx.y * 128 + wave * 16;
    v8f ai = vzero8(), af = vzero8(), ag = vzero8(), ao = vzero8();
    for (int k0 = 0; k0 < 512; k0 += 32) {
        v16bf a  = load_fragA(Hs, 512, k0, lane);
        ai = wmma_bf16(a, load_fragB(whh + (size_t)(0 * 512 + hcol0) * 512, 512, k0, lane), ai);
        af = wmma_bf16(a, load_fragB(whh + (size_t)(1 * 512 + hcol0) * 512, 512, k0, lane), af);
        ag = wmma_bf16(a, load_fragB(whh + (size_t)(2 * 512 + hcol0) * 512, 512, k0, lane), ag);
        ao = wmma_bf16(a, load_fragB(whh + (size_t)(3 * 512 + hcol0) * 512, 512, k0, lane), ao);
    }
    int half = lane >> 4, nn = lane & 15;
    int col = hcol0 + nn;
#pragma unroll
    for (int r = 0; r < 8; ++r) {
        int n = row0 + half * 8 + r;
        const float* xr = xgT + (size_t)n * xgStride;
        float gi = ai[r] + xr[col];
        float gf = af[r] + xr[512 + col];
        float gg = ag[r] + xr[1024 + col];
        float go = ao[r] + xr[1536 + col];
        size_t ci = (size_t)n * 512 + col;
        float c = sigm(gf) * cst[ci] + sigm(gi) * tanhf(gg);
        cst[ci] = c;
        float h = sigm(go) * tanhf(c);
        hnext[ci] = f2bf(h);
        houtT[(size_t)n * houtStride + col] = h;
    }
}

// ---------------- pooling / attention / scoring (small, VALU) ----------------

__global__ void k_avgpool(const float* __restrict__ in, const int* __restrict__ lens,
                          float* __restrict__ out, long total, int T, int Dd, int doTanh) {
    long i = (long)blockIdx.x * blockDim.x + threadIdx.x;
    if (i >= total) return;
    long n = i / Dd;
    int  d = (int)(i % Dd);
    int  L = lens[n];
    const float* p = in + ((size_t)n * T) * Dd + d;
    float s = 0.0f;
    for (int t = 0; t < L; ++t) s += p[(size_t)t * Dd];
    s /= (float)L;
    out[i] = doTanh ? tanhf(s) : s;
}

__global__ void k_att_scores(const float* __restrict__ c, const float* __restrict__ q,
                             const float* wcq, const float* bcq,
                             const float* wc,  const float* bc,
                             const float* wq,  const float* bq,
                             float* __restrict__ sscore) {
    int i = blockIdx.x * blockDim.x + threadIdx.x;
    if (i >= 16 * 40 * 20) return;
    int b = i / 800, rem = i % 800, ci = rem / 20, qj = rem % 20;
    const float* cp = c + ((size_t)b * 40 + ci) * 1024;
    const float* qp = q + ((size_t)b * 20 + qj) * 1024;
    float acq = 0.0f, ac = 0.0f, aq = 0.0f;
    for (int d = 0; d < 1024; ++d) {
        float cv = cp[d], qv = qp[d];
        acq += cv * qv * wcq[d];
        ac  += cv * wc[d];
        aq  += qv * wq[d];
    }
    sscore[i] = ac + bc[0] + aq + bq[0] + acq + bcq[0];
}

// one block per (b, c_row): softmax over q_len(20), produce c2q row + row max
__global__ void k_softmax_c2q(const float* __restrict__ sscore, const float* __restrict__ q,
                              float* __restrict__ c2q, float* __restrict__ smax) {
    int bi = blockIdx.x;                 // 0..639  (b*40 + i)
    int b = bi / 40;
    const float* sr = sscore + (size_t)bi * 20;
    float mx = sr[0];
    for (int j = 1; j < 20; ++j) mx = fmaxf(mx, sr[j]);
    float sum = 0.0f;
    for (int j = 0; j < 20; ++j) sum += __expf(sr[j] - mx);
    for (int d = threadIdx.x; d < 1024; d += blockDim.x) {
        float acc = 0.0f;
        for (int j = 0; j < 20; ++j)
            acc += __expf(sr[j] - mx) * q[((size_t)b * 20 + j) * 1024 + d];
        c2q[(size_t)bi * 1024 + d] = acc / sum;
    }
    if (threadIdx.x == 0) smax[bi] = mx;
}

// one block per b: softmax over c_len(40) of row-maxes, produce q2c[b, 1024]
__global__ void k_q2c(const float* __restrict__ smax, const float* __restrict__ c,
                      float* __restrict__ q2c) {
    int b = blockIdx.x;
    const float* mr = smax + (size_t)b * 40;
    float mx = mr[0];
    for (int i = 1; i < 40; ++i) mx = fmaxf(mx, mr[i]);
    float sum = 0.0f;
    for (int i = 0; i < 40; ++i) sum += __expf(mr[i] - mx);
    for (int d = threadIdx.x; d < 1024; d += blockDim.x) {
        float acc = 0.0f;
        for (int i = 0; i < 40; ++i)
            acc += __expf(mr[i] - mx) * c[((size_t)b * 40 + i) * 1024 + d];
        q2c[(size_t)b * 1024 + d] = acc / sum;
    }
}

__global__ void k_docterm(const float* __restrict__ waq, const float* __restrict__ cdoc,
                          const float* __restrict__ qdoc, float* __restrict__ docterm) {
    int i = blockIdx.x * blockDim.x + threadIdx.x;
    if (i >= 16 * 512) return;
    int b = i / 512, h = i % 512;
    const float* wr = waq + (size_t)h * 6144;
    const float* cd = cdoc + (size_t)b * 1024;
    const float* qd = qdoc + (size_t)b * 1024;
    float acc = 0.0f;
    for (int k = 0; k < 1024; ++k) acc += wr[4096 + k] * cd[k] + wr[5120 + k] * qd[k];
    docterm[i] = acc;
}

__global__ void k_hid(const float* __restrict__ c, const float* __restrict__ c2q,
                      const float* __restrict__ q2c, const float* __restrict__ waq,
                      const float* __restrict__ docterm, float* __restrict__ hid) {
    long i = (long)blockIdx.x * blockDim.x + threadIdx.x;
    if (i >= (long)16 * 40 * 512) return;
    int h  = (int)(i % 512);
    long bs = i / 512;
    int b  = (int)(bs / 40);
    const float* wr  = waq + (size_t)h * 6144;
    const float* cp  = c   + (size_t)bs * 1024;
    const float* c2p = c2q + (size_t)bs * 1024;
    const float* q2p = q2c + (size_t)b * 1024;
    float acc = docterm[(size_t)b * 512 + h];
    for (int k = 0; k < 1024; ++k) {
        float cv = cp[k], xv = c2p[k];
        acc += cv * wr[k] + xv * wr[1024 + k] + cv * xv * wr[2048 + k] + cv * q2p[k] * wr[3072 + k];
    }
    hid[i] = acc;
}

__global__ void k_vsal(const float* __restrict__ wsal, const float* __restrict__ cdoc,
                       float* __restrict__ vsal) {
    int i = blockIdx.x * blockDim.x + threadIdx.x;
    if (i >= 16 * 1024) return;
    int b = i / 1024, r = i % 1024;
    const float* wr = wsal + (size_t)r * 1024;
    const float* cd = cdoc + (size_t)b * 1024;
    float acc = 0.0f;
    for (int j = 0; j < 1024; ++j) acc += wr[j] * cd[j];
    vsal[i] = acc;
}

__global__ void k_pre(const float* __restrict__ cout, const float* __restrict__ wcont,
                      const float* __restrict__ vsal, const float* __restrict__ hid,
                      const float* __restrict__ wprob, const float* __restrict__ abspos,
                      const float* __restrict__ wabs, const int* __restrict__ nsent,
                      const float* __restrict__ biasP, float* __restrict__ pre) {
    int i = blockIdx.x * blockDim.x + threadIdx.x;
    if (i >= 16 * 40) return;
    int b = i / 40, s = i % 40;
    const float* cp = cout + (size_t)i * 1024;
    const float* vp = vsal + (size_t)b * 1024;
    float acc = biasP[0];
    for (int k = 0; k < 1024; ++k) acc += cp[k] * (wcont[k] + vp[k]);
    const float* hp = hid + (size_t)i * 512;
    for (int h = 0; h < 512; ++h) acc += hp[h] * wprob[h];
    int idx = (s <= nsent[b]) ? (s + 1) : 0;
    const float* ap = abspos + (size_t)idx * 50;
    for (int r = 0; r < 50; ++r) acc += ap[r] * wabs[r];
    pre[i] = acc;
}

// novelty scan: one block per document, sequential over the 40 sentences
__global__ void k_scan(const float* __restrict__ cout, const float* __restrict__ Wn,
                       const float* __restrict__ pre, float* __restrict__ out) {
    int b = blockIdx.x, tid = threadIdx.x;
    __shared__ float sstate[1024];
    __shared__ float ts[1024];
    __shared__ float red[256];
    __shared__ float psh;
    for (int k = tid; k < 1024; k += 256) sstate[k] = 0.0f;
    __syncthreads();
    for (int s = 0; s < 40; ++s) {
        for (int k = tid; k < 1024; k += 256) ts[k] = tanhf(sstate[k]);
        __syncthreads();
        const float* x = cout + ((size_t)b * 40 + s) * 1024;
        float acc = 0.0f;
        for (int i = tid; i < 1024; i += 256) {
            const float* wr = Wn + (size_t)i * 1024;
            float u = 0.0f;
            for (int j = 0; j < 1024; ++j) u += wr[j] * ts[j];
            acc += x[i] * u;
        }
        red[tid] = acc;
        __syncthreads();
        for (int w = 128; w > 0; w >>= 1) {
            if (tid < w) red[tid] += red[tid + w];
            __syncthreads();
        }
        if (tid == 0) {
            float p = sigm(pre[b * 40 + s] - red[0]);
            psh = p;
            out[b * 40 + s] = p;
        }
        __syncthreads();
        float p = psh;
        for (int k = tid; k < 1024; k += 256) sstate[k] += p * x[k];
        __syncthreads();
    }
}

// ---------------- host orchestration ----------------

static inline unsigned gf(long n) { return (unsigned)((n + 255) / 256); }

extern "C" void kernel_launch(void* const* d_in, const int* in_sizes, int n_in,
                              void* d_out, int out_size, void* d_ws, size_t ws_size,
                              hipStream_t stream) {
    (void)in_sizes; (void)n_in; (void)out_size; (void)ws_size;
    const int D = 16, S = 40, T = 60, SP = 20, E = 256;
    const int NC = D * S;   // 640
    const int NP = D * SP;  // 320

    const int*   tok_c   = (const int*)d_in[0];
    const int*   nsent   = (const int*)d_in[3];
    const int*   slen_c  = (const int*)d_in[4];
    const int*   tok_p   = (const int*)d_in[5];
    const int*   nsent_p = (const int*)d_in[8];
    const int*   slen_p  = (const int*)d_in[9];
    const float* Pemb    = (const float*)d_in[10];
    const float* WihFw = (const float*)d_in[11], *WhhFw = (const float*)d_in[12], *bFw = (const float*)d_in[13];
    const float* WihBw = (const float*)d_in[14], *WhhBw = (const float*)d_in[15], *bBw = (const float*)d_in[16];
    const float* WihFs = (const float*)d_in[17], *WhhFs = (const float*)d_in[18], *bFs = (const float*)d_in[19];
    const float* WihBs = (const float*)d_in[20], *WhhBs = (const float*)d_in[21], *bBs = (const float*)d_in[22];
    const float* w_c   = (const float*)d_in[23], *b_c = (const float*)d_in[24];
    const float* w_q   = (const float*)d_in[25], *b_q = (const float*)d_in[26];
    const float* w_cq  = (const float*)d_in[27], *b_cq = (const float*)d_in[28];
    const float* abspos  = (const float*)d_in[29];
    const float* w_cont  = (const float*)d_in[30];
    const float* w_sal   = (const float*)d_in[31];
    const float* w_nov   = (const float*)d_in[32];
    const float* w_aq    = (const float*)d_in[33];
    const float* w_prob  = (const float*)d_in[34];
    const float* w_abs   = (const float*)d_in[35];
    const float* biasP   = (const float*)d_in[36];

    // workspace allocator (256B aligned)
    size_t off = 0;
    char* base = (char*)d_ws;
    auto alloc = [&](size_t bytes) -> char* {
        char* p = base + off;
        off = (off + bytes + 255) & ~(size_t)255;
        return p;
    };
    bf16* bWihFw = (bf16*)alloc((size_t)2048 * 256 * 2);
    bf16* bWihBw = (bf16*)alloc((size_t)2048 * 256 * 2);
    bf16* bWhhFw = (bf16*)alloc((size_t)2048 * 512 * 2);
    bf16* bWhhBw = (bf16*)alloc((size_t)2048 * 512 * 2);
    bf16* bWihFs = (bf16*)alloc((size_t)2048 * 1024 * 2);
    bf16* bWihBs = (bf16*)alloc((size_t)2048 * 1024 * 2);
    bf16* bWhhFs = (bf16*)alloc((size_t)2048 * 512 * 2);
    bf16* bWhhBs = (bf16*)alloc((size_t)2048 * 512 * 2);
    bf16* embC   = (bf16*)alloc((size_t)NC * T * E * 2);
    bf16* embP   = (bf16*)alloc((size_t)NP * T * E * 2);
    float* xg    = (float*)alloc((size_t)NC * T * 2048 * 4);
    bf16* hA     = (bf16*)alloc((size_t)NC * 512 * 2);
    bf16* hB     = (bf16*)alloc((size_t)NC * 512 * 2);
    float* cst   = (float*)alloc((size_t)NC * 512 * 4);
    float* woutC = (float*)alloc((size_t)NC * T * 1024 * 4);
    float* woutP = (float*)alloc((size_t)NP * T * 1024 * 4);
    float* sembCf= (float*)alloc((size_t)NC * 1024 * 4);
    float* sembPf= (float*)alloc((size_t)NP * 1024 * 4);
    bf16* sembC  = (bf16*)alloc((size_t)NC * 1024 * 2);
    bf16* sembP  = (bf16*)alloc((size_t)NP * 1024 * 2);
    float* soutC = (float*)alloc((size_t)D * S * 1024 * 4);
    float* soutP = (float*)alloc((size_t)D * SP * 1024 * 4);
    float* cdoc  = (float*)alloc((size_t)D * 1024 * 4);
    float* qdoc  = (float*)alloc((size_t)D * 1024 * 4);
    float* sscore= (float*)alloc((size_t)D * S * SP * 4);
    float* smax  = (float*)alloc((size_t)D * S * 4);
    float* c2q   = (float*)alloc((size_t)D * S * 1024 * 4);
    float* q2c   = (float*)alloc((size_t)D * 1024 * 4);
    float* dterm = (float*)alloc((size_t)D * 512 * 4);
    float* hid   = (float*)alloc((size_t)D * S * 512 * 4);
    float* vsal  = (float*)alloc((size_t)D * 1024 * 4);
    float* pre   = (float*)alloc((size_t)D * S * 4);

    // 1. convert weights to bf16
    k_f2bf<<<gf(2048L * 256), 256, 0, stream>>>(WihFw, bWihFw, 2048L * 256);
    k_f2bf<<<gf(2048L * 256), 256, 0, stream>>>(WihBw, bWihBw, 2048L * 256);
    k_f2bf<<<gf(2048L * 512), 256, 0, stream>>>(WhhFw, bWhhFw, 2048L * 512);
    k_f2bf<<<gf(2048L * 512), 256, 0, stream>>>(WhhBw, bWhhBw, 2048L * 512);
    k_f2bf<<<gf(2048L * 1024), 256, 0, stream>>>(WihFs, bWihFs, 2048L * 1024);
    k_f2bf<<<gf(2048L * 1024), 256, 0, stream>>>(WihBs, bWihBs, 2048L * 1024);
    k_f2bf<<<gf(2048L * 512), 256, 0, stream>>>(WhhFs, bWhhFs, 2048L * 512);
    k_f2bf<<<gf(2048L * 512), 256, 0, stream>>>(WhhBs, bWhhBs, 2048L * 512);

    // 2. embedding gather (f32 table -> bf16 activations)
    k_embed<<<gf((long)NC * T * E), 256, 0, stream>>>(tok_c, Pemb, embC, (long)NC * T * E, E);
    k_embed<<<gf((long)NP * T * E), 256, 0, stream>>>(tok_p, Pemb, embP, (long)NP * T * E, E);

    // bidirectional LSTM runner (WMMA xg GEMM + fused WMMA recurrent steps)
    auto run_birnn = [&](const bf16* x, int N, int Tt, int K,
                         const bf16* wihF, const float* bF, const bf16* whhF,
                         const bf16* wihB, const float* bB, const bf16* whhB,
                         float* hout) {
        for (int dir = 0; dir < 2; ++dir) {
            const bf16*  wih = dir ? wihB : wihF;
            const float* bb  = dir ? bB   : bF;
            const bf16*  whh = dir ? whhB : whhF;
            k_gemm_bias<<<dim3((unsigned)(N * Tt / 16), 4), 256, 0, stream>>>(x, wih, bb, xg, K);
            k_zero<<<gf((long)N * 512), 256, 0, stream>>>(cst, (long)N * 512);
            k_zero<<<gf((long)N * 256), 256, 0, stream>>>((float*)hA, (long)N * 256);
            bf16 *hp = hA, *hn = hB;
            for (int ss = 0; ss < Tt; ++ss) {
                int t = dir ? (Tt - 1 - ss) : ss;
                k_lstm_step<<<dim3((unsigned)(N / 16), 4), 256, 0, stream>>>(
                    xg + (size_t)t * 2048, Tt * 2048, whh, hp, hn, cst,
                    hout + (size_t)t * 1024 + dir * 512, Tt * 1024);
                bf16* tmp = hp; hp = hn; hn = tmp;
            }
        }
    };

    // 3. word-level BiLSTMs
    run_birnn(embC, NC, T, 256, bWihFw, bFw, bWhhFw, bWihBw, bBw, bWhhBw, woutC);
    run_birnn(embP, NP, T, 256, bWihFw, bFw, bWhhFw, bWihBw, bBw, bWhhBw, woutP);

    // 4. sentence embeddings (masked mean over tokens) -> bf16
    k_avgpool<<<gf((long)NC * 1024), 256, 0, stream>>>(woutC, slen_c, sembCf, (long)NC * 1024, T, 1024, 0);
    k_avgpool<<<gf((long)NP * 1024), 256, 0, stream>>>(woutP, slen_p, sembPf, (long)NP * 1024, T, 1024, 0);
    k_f2bf<<<gf((long)NC * 1024), 256, 0, stream>>>(sembCf, sembC, (long)NC * 1024);
    k_f2bf<<<gf((long)NP * 1024), 256, 0, stream>>>(sembPf, sembP, (long)NP * 1024);

    // 5. sentence-level BiLSTMs (batch = D, K = 1024)
    run_birnn(sembC, D, S,  1024, bWihFs, bFs, bWhhFs, bWihBs, bBs, bWhhBs, soutC);
    run_birnn(sembP, D, SP, 1024, bWihFs, bFs, bWhhFs, bWihBs, bBs, bWhhBs, soutP);

    // 6. document embeddings (masked mean over sentences, tanh)
    k_avgpool<<<gf((long)D * 1024), 256, 0, stream>>>(soutC, nsent,   cdoc, (long)D * 1024, S,  1024, 1);
    k_avgpool<<<gf((long)D * 1024), 256, 0, stream>>>(soutP, nsent_p, qdoc, (long)D * 1024, SP, 1024, 1);

    // 7. BiDAF attention flow
    k_att_scores<<<gf(16L * 40 * 20), 256, 0, stream>>>(soutC, soutP, w_cq, b_cq, w_c, b_c, w_q, b_q, sscore);
    k_softmax_c2q<<<640, 256, 0, stream>>>(sscore, soutP, c2q, smax);
    k_q2c<<<16, 256, 0, stream>>>(smax, soutC, q2c);

    // 8. attention_and_query MLP + scoring terms
    k_docterm<<<gf(16L * 512), 256, 0, stream>>>(w_aq, cdoc, qdoc, dterm);
    k_hid<<<gf(16L * 40 * 512), 256, 0, stream>>>(soutC, c2q, q2c, w_aq, dterm, hid);
    k_vsal<<<gf(16L * 1024), 256, 0, stream>>>(w_sal, cdoc, vsal);
    k_pre<<<gf(16L * 40), 256, 0, stream>>>(soutC, w_cont, vsal, hid, w_prob, abspos, w_abs, nsent, biasP, pre);

    // 9. sequential novelty scan -> output probabilities (D, S)
    k_scan<<<16, 256, 0, stream>>>(soutC, w_nov, pre, (float*)d_out);
}
